// StreamingCodecEncoder_44753559224671
// MI455X (gfx1250) — compile-verified, matrix operand order checked by reading, not run on hardware
//
#include <hip/hip_runtime.h>
#include <hip/hip_bf16.h>
#include <stdint.h>

// ---------------------------------------------------------------------------
// StreamingCodecEncoder for MI455X (gfx1250, wave32, WMMA)
//   Prep kernels : weights -> fragment-swizzled bf16; codebooks -> bf16 + f32 norms
//   Kernel A     : conv front-end (conv0 VALU; conv1/2/3 bf16 WMMA, B-frags = 2x b128)
//   Kernel B     : RVQ, async bf16 codebook streaming, B-frags = 2x ds_load_b128
// ---------------------------------------------------------------------------

#define B_TOTAL 8192
#define D_MODEL 256
#define NCB 8
#define CBSZ 1024

typedef __attribute__((ext_vector_type(16))) __bf16 bf16x16;
typedef __attribute__((ext_vector_type(8)))  float  f32x8;
typedef __attribute__((ext_vector_type(4)))  float  f32x4;
typedef __attribute__((ext_vector_type(4)))  unsigned int u32x4;
typedef int v4i __attribute__((vector_size(16)));

union FragBF { bf16x16 v; unsigned int w[8]; u32x4 q[2]; };

// Two adjacent f32 -> packed bf16 dword (backend: v_cvt_pk_bf16_f32, RNE).
__device__ __forceinline__ unsigned int pk_bf16(float a, float b) {
  union { __bf16 h[2]; unsigned int u; } x;
  x.h[0] = (__bf16)a; x.h[1] = (__bf16)b;
  return x.u;
}

__device__ __forceinline__ f32x8 wmma_bf16(bf16x16 a, bf16x16 b, f32x8 c) {
  return __builtin_amdgcn_wmma_f32_16x16x32_bf16(false, a, false, b, (short)0, c, false, false);
}

__device__ __forceinline__ float elu1(float x) { return x > 0.f ? x : (expf(x) - 1.f); }

// k-offset (inside a 32-wide K block) of fragment element pair j (0..7),
// lane-half hv (ISA 7.12.2 16-bit A/B layout). Pair = {k, k+1}.
#define KPAIR(j, hv) ((((j) & 4) << 2) + ((hv) << 3) + (((j) & 3) << 1))

// ---------------------------------------------------------------------------
// Workspace layout (bytes) — written in full by prep kernels every call.
// ---------------------------------------------------------------------------
#define WS_CBH  0                               // bf16 codebooks: 8*1024*256*2
#define WS_NRM  (WS_CBH + NCB * CBSZ * D_MODEL * 2)        // f32 norms: 8192*4
#define WS_W1   (WS_NRM + NCB * CBSZ * 4)                  // 128*512*2
#define WS_W2   (WS_W1 + 128 * 512 * 2)                    // 256*1280*2
#define WS_W3   (WS_W2 + 256 * 1280 * 2)                   // 256*3072*2
#define WS_END  (WS_W3 + 256 * 3072 * 2)                   // ~6.59 MB total

// ---------------------------------------------------------------------------
// Prep: weights -> bf16 in exact fragment order: dword idx =
//   ((co*KB + kb)*2 + hv)*8 + j   holds  pk(W[co*K + kb*32 + KPAIR(j,hv)], +1)
// ---------------------------------------------------------------------------
__global__ void __launch_bounds__(256) swizzle_weights_kernel(
    const float* __restrict__ W, unsigned int* __restrict__ out, int N, int KB) {
  const int total = N * KB * 16;
  const int K = KB * 32;
  for (int idx = blockIdx.x * 256 + threadIdx.x; idx < total; idx += gridDim.x * 256) {
    int co = idx / (KB * 16), r = idx % (KB * 16);
    int kb = r >> 4, hv = (r >> 3) & 1, j = r & 7;
    int kk = kb * 32 + KPAIR(j, hv);
    const float* wr = W + (size_t)co * K + kk;
    out[idx] = pk_bf16(wr[0], wr[1]);
  }
}

// Prep: codebooks -> row-contiguous bf16 (128 dwords per code).
__global__ void __launch_bounds__(256) cvt_codebook_kernel(
    const float* __restrict__ cb, unsigned int* __restrict__ cbh) {
  const int total = NCB * CBSZ * 128;
  for (int idx = blockIdx.x * 256 + threadIdx.x; idx < total; idx += gridDim.x * 256) {
    int code = idx >> 7, dp = (idx & 127) << 1;
    const float* p = cb + (size_t)code * D_MODEL + dp;
    cbh[idx] = pk_bf16(p[0], p[1]);
  }
}

// Prep: exact f32 ||c||^2 per code.
__global__ void __launch_bounds__(256) codebook_norms_kernel(
    const float* __restrict__ cb, float* __restrict__ nrm) {
  int code = blockIdx.x * 256 + threadIdx.x;
  if (code < NCB * CBSZ) {
    const float* p = cb + (size_t)code * D_MODEL;
    float s = 0.f;
    for (int d = 0; d < D_MODEL; ++d) s += p[d] * p[d];
    nrm[code] = s;
  }
}

// ---------------------------------------------------------------------------
// Kernel A: encoder front-end.  One block = 4 batch elements, 8 waves.
// ---------------------------------------------------------------------------
#define BT 4
#define LA 242
#define T0 120
#define LY0 124
#define T1 30
#define LY1 35
#define T2 6
#define LY2 12

#define OFF_AUD 0
#define SZ_AUD  (BT * LA)
#define OFF_Y0  (OFF_AUD + SZ_AUD)
#define SZ_Y0   (BT * 64 * LY0)
#define OFF_Y1  (OFF_Y0 + SZ_Y0)
#define SZ_Y1   (BT * 128 * LY1)
#define OFF_Y2  (OFF_Y1 + SZ_Y1)
#define SZ_Y2   (BT * 256 * LY2)
#define SMEM_A_FLOATS (OFF_Y2 + SZ_Y2)   // 62920 floats = 251680 B

__global__ void __launch_bounds__(256) encoder_kernel(
    const float* __restrict__ audio, const float* __restrict__ state,
    const float* __restrict__ W0, const float* __restrict__ b0,
    const unsigned int* __restrict__ W1s, const float* __restrict__ b1,
    const unsigned int* __restrict__ W2s, const float* __restrict__ b2,
    const unsigned int* __restrict__ W3s, const float* __restrict__ b3,
    float* __restrict__ dout)
{
  extern __shared__ float smem[];
  const int tid  = threadIdx.x;
  const int lane = tid & 31;
  const int wv   = tid >> 5;
  const int hv   = lane >> 4;
  const int ln   = lane & 15;
  const int bbase = blockIdx.x * BT;

  float* s_aud = smem + OFF_AUD;
  float* s_y0  = smem + OFF_Y0;
  float* s_y1  = smem + OFF_Y1;
  float* s_y2  = smem + OFF_Y2;
  float* out_state = dout + (size_t)B_TOTAL * NCB;

  // Zero LDS: provides all causal left-padding for free.
  for (int i = tid; i < SMEM_A_FLOATS; i += 256) smem[i] = 0.f;

  // new_state[:, :, 0:3] = state_in[:, :, 1:4]   (roll left)
  for (int i = tid; i < BT * D_MODEL * 3; i += 256) {
    int e = i / (D_MODEL * 3), r = i % (D_MODEL * 3);
    int d = r / 3, t = r % 3;
    size_t o = (size_t)(bbase + e) * (D_MODEL * 4) + (size_t)d * 4;
    out_state[o + t] = state[o + t + 1];
  }
  __syncthreads();

  // Stage audio into LDS (offset 2 = causal pad for k4 s2).
  for (int i = tid; i < BT * 240; i += 256) {
    int e = i / 240, t = i % 240;
    s_aud[e * LA + 2 + t] = audio[(size_t)(bbase + e) * 240 + t];
  }
  __syncthreads();

  // conv0: cin=1, k=4, s=2  (+ELU) — tiny, plain VALU.
  for (int i = tid; i < BT * 64 * T0; i += 256) {
    int e = i / (64 * T0), r = i % (64 * T0);
    int c = r / T0, t = r % T0;
    float acc = b0[c];
    #pragma unroll
    for (int j = 0; j < 4; ++j) acc += W0[c * 4 + j] * s_aud[e * LA + 2 * t + j];
    s_y0[(e * 64 + c) * LY0 + 4 + t] = elu1(acc);
  }
  __syncthreads();

  // conv1 as GEMM: M=BT*30=120 rows, N=128, K=512 (KB=16). Wave w owns M-tile w.
  {
    const f32x8 VZ = {0.f,0.f,0.f,0.f,0.f,0.f,0.f,0.f};
    f32x8 acc[8];
    #pragma unroll
    for (int nt = 0; nt < 8; ++nt) acc[nt] = VZ;
    const int mrow = wv * 16 + ln;
    const bool av  = (mrow < BT * T1);
    const int mr   = av ? mrow : 0;
    const int e1 = mr / T1, t1 = mr % T1;
    const float* arow = &s_y0[e1 * 64 * LY0];
    #pragma unroll 1
    for (int kb = 0; kb < 16; ++kb) {
      FragBF a;
      {
        // 8-run stays inside one ci; 16B-aligned (LY0, 4*t1 multiples of 4)
        int klo = kb * 32 + hv * 8;
        const float* p0 = arow + (klo >> 3) * LY0 + 4 * t1;        // k = klo..klo+7
        const float* p1 = arow + ((klo + 16) >> 3) * LY0 + 4 * t1; // k = klo+16..+23
        f32x4 a0 = *(const f32x4*)(p0), a1 = *(const f32x4*)(p0 + 4);
        f32x4 a2 = *(const f32x4*)(p1), a3 = *(const f32x4*)(p1 + 4);
        a.w[0] = av ? pk_bf16(a0[0], a0[1]) : 0u;
        a.w[1] = av ? pk_bf16(a0[2], a0[3]) : 0u;
        a.w[2] = av ? pk_bf16(a1[0], a1[1]) : 0u;
        a.w[3] = av ? pk_bf16(a1[2], a1[3]) : 0u;
        a.w[4] = av ? pk_bf16(a2[0], a2[1]) : 0u;
        a.w[5] = av ? pk_bf16(a2[2], a2[3]) : 0u;
        a.w[6] = av ? pk_bf16(a3[0], a3[1]) : 0u;
        a.w[7] = av ? pk_bf16(a3[2], a3[3]) : 0u;
      }
      #pragma unroll
      for (int nt = 0; nt < 8; ++nt) {
        FragBF bb;
        const unsigned int* p = W1s + ((((nt * 16 + ln) * 16 + kb) * 2 + hv) << 3);
        bb.q[0] = *(const u32x4*)(p);
        bb.q[1] = *(const u32x4*)(p + 4);
        acc[nt] = wmma_bf16(a.v, bb.v, acc[nt]);
      }
    }
    #pragma unroll
    for (int nt = 0; nt < 8; ++nt) {
      int co = nt * 16 + ln;
      #pragma unroll
      for (int v = 0; v < 8; ++v) {
        int row = wv * 16 + v + hv * 8;
        if (row < BT * T1) {
          int e = row / T1, t = row % T1;
          s_y1[(e * 128 + co) * LY1 + 5 + t] = elu1(acc[nt][v] + b1[co]);
        }
      }
    }
  }
  __syncthreads();

  // conv2 as GEMM: M=24 rows (2 tiles), N=256 (16 tiles), K=1280 (KB=40).
  {
    const f32x8 VZ = {0.f,0.f,0.f,0.f,0.f,0.f,0.f,0.f};
    f32x8 acc[4];
    #pragma unroll
    for (int nn = 0; nn < 4; ++nn) acc[nn] = VZ;
    const int mt = wv & 1, ntb = (wv >> 1) * 4;
    const int mrow = mt * 16 + ln;
    const bool av  = (mrow < BT * T2);
    const int mr   = av ? mrow : 0;
    const int e2 = mr / T2, t2 = mr % T2;
    const float* arow = &s_y1[e2 * 128 * LY1];
    #pragma unroll 1
    for (int kb = 0; kb < 40; ++kb) {
      FragBF a;
      #pragma unroll
      for (int j = 0; j < 8; ++j) {
        int kk = kb * 32 + KPAIR(j, hv);
        int ci = kk / 10;
        int off = ci * LY1 + 5 * t2 + (kk - ci * 10);    // even k: pair in-row
        a.w[j] = pk_bf16(av ? arow[off] : 0.f, av ? arow[off + 1] : 0.f);
      }
      #pragma unroll
      for (int nn = 0; nn < 4; ++nn) {
        FragBF bb;
        const unsigned int* p = W2s + (((((ntb + nn) * 16 + ln) * 40 + kb) * 2 + hv) << 3);
        bb.q[0] = *(const u32x4*)(p);
        bb.q[1] = *(const u32x4*)(p + 4);
        acc[nn] = wmma_bf16(a.v, bb.v, acc[nn]);
      }
    }
    #pragma unroll
    for (int nn = 0; nn < 4; ++nn) {
      int co = (ntb + nn) * 16 + ln;
      #pragma unroll
      for (int v = 0; v < 8; ++v) {
        int row = mt * 16 + v + hv * 8;
        if (row < BT * T2) {
          int e = row / T2, t = row % T2;
          s_y2[(e * 256 + co) * LY2 + 6 + t] = elu1(acc[nn][v] + b2[co]);
        }
      }
    }
  }
  __syncthreads();

  // conv3 as GEMM: M=4 rows (1 tile), N=256 (16 tiles), K=3072 (KB=96). No ELU.
  {
    const f32x8 VZ = {0.f,0.f,0.f,0.f,0.f,0.f,0.f,0.f};
    f32x8 acc[2];
    acc[0] = VZ; acc[1] = VZ;
    const int mrow = ln;
    const bool av  = (mrow < BT);
    const int mr   = av ? mrow : 0;
    const float* arow = &s_y2[mr * 256 * LY2];
    #pragma unroll 1
    for (int kb = 0; kb < 96; ++kb) {
      FragBF a;
      #pragma unroll
      for (int j = 0; j < 8; ++j) {
        int kk = kb * 32 + KPAIR(j, hv);
        int ci = kk / 12;
        int off = ci * LY2 + (kk - ci * 12);             // even k: pair in-row
        a.w[j] = pk_bf16(av ? arow[off] : 0.f, av ? arow[off + 1] : 0.f);
      }
      #pragma unroll
      for (int nn = 0; nn < 2; ++nn) {
        FragBF bb;
        const unsigned int* p = W3s + (((((wv * 2 + nn) * 16 + ln) * 96 + kb) * 2 + hv) << 3);
        bb.q[0] = *(const u32x4*)(p);
        bb.q[1] = *(const u32x4*)(p + 4);
        acc[nn] = wmma_bf16(a.v, bb.v, acc[nn]);
      }
    }
    // x -> new_state[:, :, 3]  (also consumed by the RVQ kernel)
    #pragma unroll
    for (int nn = 0; nn < 2; ++nn) {
      int co = (wv * 2 + nn) * 16 + ln;
      #pragma unroll
      for (int v = 0; v < 8; ++v) {
        int row = v + hv * 8;
        if (row < BT) {
          float val = acc[nn][v] + b3[co];
          out_state[(size_t)(bbase + row) * (D_MODEL * 4) + (size_t)co * 4 + 3] = val;
        }
      }
    }
  }
}

// ---------------------------------------------------------------------------
// Kernel B: RVQ. One block = 128 batch rows (8 waves x 16 rows).
// 16 double-buffered async chunks of 64 bf16 codes per codebook; one barrier
// per chunk; B-fragments = 2x ds_load_b128; norms pre-staged per codebook.
// ---------------------------------------------------------------------------
#define MR 128
#define RST 260               // f32 stride: conflict-free + 16B aligned
#define CBHST 264             // bf16 (ushort) stride: conflict-free, 16B-aligned rows
#define CHUNK 64
#define NCHUNK 16

#define OFF_R    0
#define SZ_R     (MR * RST)                       // 33280
#define OFF_QNRM (OFF_R + SZ_R)
#define SZ_QNRM  (CBSZ)                           // 1024
#define OFF_WIDX (OFF_QNRM + SZ_QNRM)
#define SZ_WIDX  (MR)                             // 128 (ints)
#define OFF_CBH  (OFF_WIDX + SZ_WIDX)
#define SZ_CBH   (2 * CHUNK * CBHST / 2)          // 16896 floats (bf16 x 33792)
#define SMEM_B_FLOATS (OFF_CBH + SZ_CBH)          // 51328 floats = 205312 B

#if defined(__has_builtin)
#if __has_builtin(__builtin_amdgcn_global_load_async_to_lds_b128) && \
    __has_builtin(__builtin_amdgcn_s_wait_asynccnt)
#define USE_ASYNC_LDS 1
#endif
#endif

__device__ __forceinline__ void prefetch_chunk(const unsigned short* __restrict__ src,
                                               unsigned short* dstbuf, int tid) {
#ifdef USE_ASYNC_LDS
  #pragma unroll
  for (int i = 0; i < 8; ++i) {                    // 256 thr x 8 x 16B = 32KB
    int g = i * 256 + tid;
    int code = g >> 5, unit = g & 31;              // 32 x 16B units per bf16 code row
    __builtin_amdgcn_global_load_async_to_lds_b128(
        (__attribute__((address_space(1))) v4i*)(src + code * 256 + unit * 8),
        (__attribute__((address_space(3))) v4i*)(dstbuf + code * CBHST + unit * 8),
        0, 0);
  }
#else
  for (int i = tid; i < CHUNK * 128; i += 256) {
    int code = i >> 7, u = i & 127;
    ((unsigned int*)(dstbuf + code * CBHST))[u] = ((const unsigned int*)(src + code * 256))[u];
  }
#endif
}

__device__ __forceinline__ void wait_async_done() {
#ifdef USE_ASYNC_LDS
  __builtin_amdgcn_s_wait_asynccnt(0);
#endif
}

__global__ void __launch_bounds__(256) rvq_kernel(
    const unsigned short* __restrict__ cbh_g,   // bf16 codebooks (prep)
    const float* __restrict__ nrm_g,            // exact f32 norms (prep)
    const float* __restrict__ cb,               // original f32 codebooks
    float* __restrict__ dout) {
  extern __shared__ float smem[];
  float* s_r    = smem + OFF_R;
  float* s_qnrm = smem + OFF_QNRM;
  int*   s_widx = (int*)(smem + OFF_WIDX);
  unsigned short* s_cbh = (unsigned short*)(smem + OFF_CBH);

  const int tid  = threadIdx.x;
  const int lane = tid & 31;
  const int wv   = tid >> 5;
  const int hv   = lane >> 4;
  const int ln   = lane & 15;
  const int rowbase = blockIdx.x * MR;

  const float* xsrc = dout + (size_t)B_TOTAL * NCB;   // new_state[:, :, 3]
  float* tok = dout;

  // residual r0 = x  (strided gather from new_state column 3)
  for (int i = tid; i < MR * D_MODEL; i += 256) {
    int row = i >> 8, d = i & 255;
    s_r[row * RST + d] = xsrc[(size_t)(rowbase + row) * (D_MODEL * 4) + (size_t)d * 4 + 3];
  }
  __syncthreads();

  for (int q = 0; q < NCB; ++q) {
    const float* cbq = cb + (size_t)q * CBSZ * D_MODEL;
    const unsigned short* cbhq = cbh_g + (size_t)q * CBSZ * D_MODEL;

    // stage this codebook's norms (visible after first chunk barrier)
    for (int i = tid; i < CBSZ; i += 256) s_qnrm[i] = nrm_g[q * CBSZ + i];

    // A fragments for this wave's 16 rows (K=256 -> 8 k-blocks), bf16.
    FragBF afrag[8];
    {
      const float* rrow = &s_r[(wv * 16 + ln) * RST];
      #pragma unroll
      for (int kb = 0; kb < 8; ++kb) {
        const float* p0 = rrow + kb * 32 + hv * 8;
        f32x4 a0 = *(const f32x4*)(p0),      a1 = *(const f32x4*)(p0 + 4);
        f32x4 a2 = *(const f32x4*)(p0 + 16), a3 = *(const f32x4*)(p0 + 20);
        afrag[kb].w[0] = pk_bf16(a0[0], a0[1]);
        afrag[kb].w[1] = pk_bf16(a0[2], a0[3]);
        afrag[kb].w[2] = pk_bf16(a1[0], a1[1]);
        afrag[kb].w[3] = pk_bf16(a1[2], a1[3]);
        afrag[kb].w[4] = pk_bf16(a2[0], a2[1]);
        afrag[kb].w[5] = pk_bf16(a2[2], a2[3]);
        afrag[kb].w[6] = pk_bf16(a3[0], a3[1]);
        afrag[kb].w[7] = pk_bf16(a3[2], a3[3]);
      }
    }

    float mval[8]; int midx[8];
    #pragma unroll
    for (int v = 0; v < 8; ++v) { mval[v] = 3.4e38f; midx[v] = 0; }

    prefetch_chunk(cbhq, s_cbh, tid);                // chunk 0 -> buf 0
    #pragma unroll 1
    for (int c = 0; c < NCHUNK; ++c) {
      wait_async_done();
      __syncthreads();                               // chunk c resident for all
      const int bsel = c & 1;
      unsigned short* cbhbuf = s_cbh + bsel * (CHUNK * CBHST);

      if (c + 1 < NCHUNK)
        prefetch_chunk(cbhq + (size_t)(c + 1) * CHUNK * D_MODEL,
                       s_cbh + (bsel ^ 1) * (CHUNK * CBHST), tid);

      #pragma unroll
      for (int nt = 0; nt < 4; ++nt) {
        const f32x8 VZ = {0.f,0.f,0.f,0.f,0.f,0.f,0.f,0.f};
        f32x8 acc = VZ;
        const unsigned short* crow = cbhbuf + (nt * 16 + ln) * CBHST;
        #pragma unroll
        for (int kb = 0; kb < 8; ++kb) {
          FragBF bb;
          bb.q[0] = *(const u32x4*)(crow + kb * 32 + hv * 8);
          bb.q[1] = *(const u32x4*)(crow + kb * 32 + 16 + hv * 8);
          acc = wmma_bf16(afrag[kb].v, bb.v, acc);
        }
        int n0 = c * CHUNK + nt * 16 + ln;
        float nv = s_qnrm[n0];
        #pragma unroll
        for (int v = 0; v < 8; ++v) {
          float dist = nv - 2.f * acc[v];
          if (dist < mval[v]) { mval[v] = dist; midx[v] = n0; }
        }
      }
      // next iteration's entry barrier orders these reads before buffer reuse
    }

    // argmin across the 16 lanes of each half (tie-break: lowest index)
    #pragma unroll
    for (int m = 1; m < 16; m <<= 1) {
      #pragma unroll
      for (int v = 0; v < 8; ++v) {
        float ov = __shfl_xor(mval[v], m, 32);
        int   oi = __shfl_xor(midx[v], m, 32);
        if (ov < mval[v] || (ov == mval[v] && oi < midx[v])) { mval[v] = ov; midx[v] = oi; }
      }
    }
    if (ln == 0) {
      #pragma unroll
      for (int v = 0; v < 8; ++v) {
        int row = wv * 16 + hv * 8 + v;
        s_widx[row] = midx[v];
        tok[(size_t)(rowbase + row) * NCB + q] = (float)midx[v];  // idx<1024: exact
      }
    }
    __syncthreads();

    // exact f32 residual update from the (L2-resident) f32 codebook
    for (int i = tid; i < MR * D_MODEL; i += 256) {
      int row = i >> 8, d = i & 255;
      s_r[row * RST + d] -= cbq[(size_t)s_widx[row] * D_MODEL + d];
    }
    __syncthreads();
  }
}

// ---------------------------------------------------------------------------
extern "C" void kernel_launch(void* const* d_in, const int* in_sizes, int n_in,
                              void* d_out, int out_size, void* d_ws, size_t ws_size,
                              hipStream_t stream) {
  (void)in_sizes; (void)n_in; (void)out_size; (void)ws_size;
  const float* audio = (const float*)d_in[0];
  const float* state = (const float*)d_in[1];
  const float* W0 = (const float*)d_in[2];
  const float* b0 = (const float*)d_in[3];
  const float* W1 = (const float*)d_in[4];
  const float* b1 = (const float*)d_in[5];
  const float* W2 = (const float*)d_in[6];
  const float* b2 = (const float*)d_in[7];
  const float* W3 = (const float*)d_in[8];
  const float* b3 = (const float*)d_in[9];
  const float* cb = (const float*)d_in[10];
  float* out = (float*)d_out;   // [B*8 tokens (as float)] ++ [B*256*4 new_state]

  char* ws = (char*)d_ws;
  unsigned short* cbh_g = (unsigned short*)(ws + WS_CBH);
  float*          nrm_g = (float*)(ws + WS_NRM);
  unsigned int*   W1s   = (unsigned int*)(ws + WS_W1);
  unsigned int*   W2s   = (unsigned int*)(ws + WS_W2);
  unsigned int*   W3s   = (unsigned int*)(ws + WS_W3);

  // Prep (cheap, once per call): bf16 weights in fragment order, bf16 codebooks,
  // exact f32 code norms.
  swizzle_weights_kernel<<<(128 * 16 * 16 + 255) / 256, 256, 0, stream>>>(W1, W1s, 128, 16);
  swizzle_weights_kernel<<<(256 * 40 * 16 + 255) / 256, 256, 0, stream>>>(W2, W2s, 256, 40);
  swizzle_weights_kernel<<<(256 * 96 * 16 + 255) / 256, 256, 0, stream>>>(W3, W3s, 256, 96);
  cvt_codebook_kernel<<<(NCB * CBSZ * 128 + 255) / 256, 256, 0, stream>>>(cb, (unsigned int*)cbh_g);
  codebook_norms_kernel<<<(NCB * CBSZ + 255) / 256, 256, 0, stream>>>(cb, nrm_g);

  encoder_kernel<<<B_TOTAL / BT, 256, SMEM_A_FLOATS * sizeof(float), stream>>>(
      audio, state, W0, b0, W1s, b1, W2s, b2, W3s, b3, out);
  rvq_kernel<<<B_TOTAL / MR, 256, SMEM_B_FLOATS * sizeof(float), stream>>>(cbh_g, nrm_g, cb, out);
}
